// Attention_12292196401836
// MI455X (gfx1250) — matrix operand order repaired and checked by reading, not compiled, not run on hardware
//
#include <hip/hip_runtime.h>

// Problem constants (from reference)
#define NB   64      // batch
#define LL   258     // source length
#define HH   1024    // hidden
#define QQ   256     // queries
#define NQ   (NB*QQ) // 16384
#define LTILES 17            // ceil(258/16)
#define LPAD   (LTILES*16)   // 272 (score staging width)
#define LP     288           // 9*32, K-padded L for context GEMM

typedef __bf16 bf16_t;
typedef __attribute__((ext_vector_type(16))) __bf16 v16bf;
typedef __attribute__((ext_vector_type(8)))  __bf16 v8bf;
typedef __attribute__((ext_vector_type(8)))  float  v8f;

// ---------------- WMMA fragment helpers (CDNA5 wave32 layouts) ----------------

// A-matrix 16x32 bf16, row-major source. Lane holds row M=lane&15.
// Lanes 0-15: K = 0..15 ; lanes 16-31: K = 16..31 (contiguous).
__device__ __forceinline__ v16bf load_a_frag(const bf16_t* row, int lane) {
    const int k0 = ((lane >> 4) & 1) * 16;
    v8bf lo = *(const v8bf*)(row + k0);
    v8bf hi = *(const v8bf*)(row + k0 + 8);
    v16bf r;
#pragma unroll
    for (int i = 0; i < 8; ++i) { r[i] = lo[i]; r[i + 8] = hi[i]; }
    return r;
}

// B-matrix 32x16 bf16: lane holds column N=lane&15; lanes 0-15 K=0..15,
// lanes 16-31 K=16..31 (contiguous). `col` points at this column's K-run.
__device__ __forceinline__ v16bf load_b_frag(const bf16_t* col, int lane) {
    return *(const v16bf*)(col + ((lane >> 4) & 1) * 16);
}

__device__ __forceinline__ v8f wmma_bf16(v16bf a, v16bf b, v8f c) {
    return __builtin_amdgcn_wmma_f32_16x16x32_bf16(false, a, false, b,
                                                   (short)0, c, false, false);
}

// D 16x16 f32 frag: VGPR r -> M = r + 8*(lane>=16), N = lane&15
__device__ __forceinline__ void store_d_bf16(bf16_t* base, int ldm, int lane, v8f c) {
    const int nn = lane & 15, mh = ((lane >> 4) & 1) * 8;
#pragma unroll
    for (int r = 0; r < 8; ++r) base[(size_t)(mh + r) * ldm + nn] = (bf16_t)c[r];
}

// Low 32 bits of a flat LDS address are the LDS byte offset (ISA 10.2).
__device__ __forceinline__ uint32_t lds_byte_off(const void* p) {
    return (uint32_t)(uintptr_t)p;
}

// Async global->LDS 16-byte transfer (per-lane), tracked by ASYNCcnt.
__device__ __forceinline__ void async_g2l_b128(uint32_t lds_dst, const void* gsrc) {
    uint64_t ga = (uint64_t)(uintptr_t)gsrc;
    asm volatile("global_load_async_to_lds_b128 %0, %1, off"
                 :: "v"(lds_dst), "v"(ga) : "memory");
}
__device__ __forceinline__ void wait_async0() {
    asm volatile("s_wait_asynccnt 0" ::: "memory");
}

// ---------------- Pack / layout kernels ----------------

__global__ void k_cvt_bf16(const float* __restrict__ in, bf16_t* __restrict__ out, int n) {
    int i = blockIdx.x * blockDim.x + threadIdx.x;
    if (i < n) out[i] = (bf16_t)in[i];
}

// WaT[n*H + h] = W_a[h*H + n]  (so B-columns of GEMM1 are contiguous rows)
__global__ void k_waT(const float* __restrict__ Wa, bf16_t* __restrict__ out) {
    int i = blockIdx.x * blockDim.x + threadIdx.x;
    if (i < HH * HH) {
        int nn = i / HH, hh = i % HH;
        out[i] = (bf16_t)Wa[hh * HH + nn];
    }
}

// encT[((n*H)+h)*LP + l] = enc[(n*L + l)*H + h], zero-padded l in [L, LP)
__global__ void k_encT(const float* __restrict__ enc, bf16_t* __restrict__ out) {
    int i = blockIdx.x * blockDim.x + threadIdx.x;
    if (i < NB * HH * LP) {
        int l = i % LP;
        int h = (i / LP) % HH;
        int n = i / (LP * HH);
        float v = (l < LL) ? enc[((size_t)n * LL + l) * HH + h] : 0.0f;
        out[i] = (bf16_t)v;
    }
}

// ---------------- GEMM1: T = h_t @ W_a ----------------
// Block = 4 waves, 256(M) x 64(N) tile. Wave w owns M rows [w*64, w*64+64).
// B tile (64 cols x 32 K = 4KB) double-buffered in LDS via async loads:
// tile k+1 is fetched (ASYNCcnt) while tile k feeds 16 WMMAs per wave.

__global__ void __launch_bounds__(128) k_gemm_TWa(const bf16_t* __restrict__ A,   // [NQ,H]
                                                  const bf16_t* __restrict__ BT,  // [H,H] = Wa^T
                                                  bf16_t* __restrict__ T) {       // [NQ,H]
    __shared__ __align__(32) bf16_t sB[2][64][32];   // 2 x 4 KB ping-pong

    const int lane = threadIdx.x & 31;
    const int wave = threadIdx.x >> 5;
    const int mt = blockIdx.x >> 4;                  // NQ/256 = 64
    const int nt = blockIdx.x & 15;                  // H/64   = 16
    const int mBase = mt * 256 + wave * 64;
    const int nBase = nt * 64;

    // this thread's staging slice: 2 x 16B chunks of the 64x32 tile
    const int fl0 = threadIdx.x * 2;
    const int c0 = fl0 >> 2, ch0 = (fl0 & 3);
    const int c1 = (fl0 + 1) >> 2, ch1 = ((fl0 + 1) & 3);
    const bf16_t* bt0 = BT + (size_t)(nBase + c0) * HH + ch0 * 8;
    const bf16_t* bt1 = BT + (size_t)(nBase + c1) * HH + ch1 * 8;
    const uint32_t ld0 = lds_byte_off(&sB[0][0][0]) + (uint32_t)(c0 * 64 + ch0 * 16);
    const uint32_t ld1 = lds_byte_off(&sB[0][0][0]) + (uint32_t)(c1 * 64 + ch1 * 16);

    const bf16_t* arow[4];
#pragma unroll
    for (int s = 0; s < 4; ++s)
        arow[s] = A + (size_t)(mBase + s * 16 + (lane & 15)) * HH;

    v8f acc[4][4];
#pragma unroll
    for (int s = 0; s < 4; ++s)
#pragma unroll
        for (int t2 = 0; t2 < 4; ++t2) acc[s][t2] = (v8f){};

    // prologue: fill buffer 0 with k-step 0
    async_g2l_b128(ld0, bt0);
    async_g2l_b128(ld1, bt1);

    for (int kb = 0; kb < HH / 32; ++kb) {
        const int cur = kb & 1;
        wait_async0();          // my async writes into sB[cur] have landed
        __syncthreads();        // everyone's have; sB[1-cur] no longer being read
        if (kb + 1 < HH / 32) { // prefetch next tile into the other buffer
            const uint32_t bofs = (uint32_t)((1 - cur) * 64 * 32 * 2);
            async_g2l_b128(ld0 + bofs, bt0 + (kb + 1) * 32);
            async_g2l_b128(ld1 + bofs, bt1 + (kb + 1) * 32);
        }
        v16bf a[4], b[4];
#pragma unroll
        for (int s = 0; s < 4; ++s) a[s] = load_a_frag(arow[s] + kb * 32, lane);
#pragma unroll
        for (int t2 = 0; t2 < 4; ++t2)
            b[t2] = load_b_frag(&sB[cur][t2 * 16 + (lane & 15)][0], lane);
#pragma unroll
        for (int s = 0; s < 4; ++s)
#pragma unroll
            for (int t2 = 0; t2 < 4; ++t2) acc[s][t2] = wmma_bf16(a[s], b[t2], acc[s][t2]);
    }
#pragma unroll
    for (int s = 0; s < 4; ++s)
#pragma unroll
        for (int t2 = 0; t2 < 4; ++t2)
            store_d_bf16(T + (size_t)(mBase + s * 16) * HH + nBase + t2 * 16,
                         HH, lane, acc[s][t2]);
}

// ---------------- Fused attention: score -> mask/softmax/gauss -> context ----------------

__global__ void __launch_bounds__(128) k_attention(const bf16_t* __restrict__ Tm,   // [NQ,H]
                                                   const bf16_t* __restrict__ encB, // [N,L,H]
                                                   const bf16_t* __restrict__ encT, // [N,H,LP]
                                                   const int*    __restrict__ src_len,
                                                   const float*  __restrict__ p_t,
                                                   bf16_t* __restrict__ ctx) {      // [NQ,H]
    __shared__ __align__(32) float  sS[16][LPAD];   // scores, 17.4 KB
    __shared__ __align__(32) bf16_t sA[16][LP];     // align weights, 9.2 KB

    const int lane = threadIdx.x & 31;
    const int wave = threadIdx.x >> 5;
    const int n  = blockIdx.x >> 4;                 // 64 batches
    const int qt = blockIdx.x & 15;                 // Q/16 = 16 q-tiles
    const int qrow0 = n * QQ + qt * 16;
    const bf16_t* arow = Tm + (size_t)(qrow0 + (lane & 15)) * HH;

    // ---- scores: waves split L-tiles; WMMA with full EXEC, OOB columns get zero B ----
    for (int lt = wave; lt < LTILES; lt += 4) {
        const int l = lt * 16 + (lane & 15);
        const bf16_t* brow = encB + ((size_t)n * LL + (l < LL ? l : 0)) * HH;
        v8f c = {};
        for (int kb = 0; kb < HH / 32; ++kb) {
            v16bf a = load_a_frag(arow + kb * 32, lane);
            v16bf b = {};
            if (l < LL) b = load_b_frag(brow + kb * 32, lane);
            c = wmma_bf16(a, b, c);
        }
        const int nn = lane & 15, mh = ((lane >> 4) & 1) * 8;
#pragma unroll
        for (int r = 0; r < 8; ++r) sS[mh + r][lt * 16 + nn] = c[r];
    }
    __syncthreads();

    // ---- mask + softmax + gaussian (one thread per query row) ----
    const float pt   = p_t[n];
    const float send = (float)src_len[n];
    const float a0 = fmaxf(pt - 16.0f, 0.0f);
    const float a1 = fminf(pt + 16.0f, send);
    if (threadIdx.x < 16) {
        const int row = threadIdx.x;
        float mx = -3.0e38f;
        for (int l = 0; l < LL; ++l) {
            float idx = (float)l;
            float s = (idx < a0 || idx > a1) ? -1.0e9f : sS[row][l];
            sS[row][l] = s;
            mx = fmaxf(mx, s);
        }
        float sum = 0.0f;
        for (int l = 0; l < LL; ++l) {
            float e = __expf(sS[row][l] - mx);
            sS[row][l] = e;
            sum += e;
        }
        const float inv = 1.0f / sum;
        for (int l = 0; l < LL; ++l) {
            float d = (float)l - pt;
            float g = __expf(-(d * d) * (1.0f / 128.0f));
            sA[row][l] = (bf16_t)(sS[row][l] * inv * g);
        }
        for (int l = LL; l < LP; ++l) sA[row][l] = (bf16_t)0.0f;
    }
    __syncthreads();

    // ---- context: waves split H-tiles; A from LDS, B from encT ----
    for (int ht = wave; ht < HH / 16; ht += 4) {
        const bf16_t* brow = encT + ((size_t)n * HH + ht * 16 + (lane & 15)) * LP;
        v8f c = {};
#pragma unroll
        for (int lt = 0; lt < LP / 32; ++lt) {
            v16bf a = load_a_frag(&sA[lane & 15][lt * 32], lane);
            v16bf b = load_b_frag(brow + lt * 32, lane);
            c = wmma_bf16(a, b, c);
        }
        store_d_bf16(ctx + (size_t)qrow0 * HH + ht * 16, HH, lane, c);
    }
}

// ---------------- GEMM4: out = tanh(ctx @ Wc1^T + h_t @ Wc2^T) ----------------
// Same blocking as GEMM1; K runs over 2H in two phases (A = ctx, then A = h_t),
// with double-buffered async B staging across all 64 k-steps.

__global__ void __launch_bounds__(128) k_out(const bf16_t* __restrict__ ctx, // [NQ,H]
                                             const bf16_t* __restrict__ hT,  // [NQ,H]
                                             const bf16_t* __restrict__ Wc,  // [H,2H]
                                             float* __restrict__ out) {      // [NQ,H]
    __shared__ __align__(32) bf16_t sB[2][64][32];

    const int lane = threadIdx.x & 31;
    const int wave = threadIdx.x >> 5;
    const int mt = blockIdx.x >> 4;                  // NQ/256 = 64
    const int nt = blockIdx.x & 15;                  // H/64   = 16
    const int mBase = mt * 256 + wave * 64;
    const int nBase = nt * 64;

    const int fl0 = threadIdx.x * 2;
    const int c0 = fl0 >> 2, ch0 = (fl0 & 3);
    const int c1 = (fl0 + 1) >> 2, ch1 = ((fl0 + 1) & 3);
    const bf16_t* bt0 = Wc + (size_t)(nBase + c0) * (2 * HH) + ch0 * 8;
    const bf16_t* bt1 = Wc + (size_t)(nBase + c1) * (2 * HH) + ch1 * 8;
    const uint32_t ld0 = lds_byte_off(&sB[0][0][0]) + (uint32_t)(c0 * 64 + ch0 * 16);
    const uint32_t ld1 = lds_byte_off(&sB[0][0][0]) + (uint32_t)(c1 * 64 + ch1 * 16);

    v8f acc[4][4];
#pragma unroll
    for (int s = 0; s < 4; ++s)
#pragma unroll
        for (int t2 = 0; t2 < 4; ++t2) acc[s][t2] = (v8f){};

    async_g2l_b128(ld0, bt0);
    async_g2l_b128(ld1, bt1);

    const int KB = 2 * HH / 32;                      // 64 k-steps over concat dim
    for (int kb = 0; kb < KB; ++kb) {
        const int cur = kb & 1;
        wait_async0();
        __syncthreads();
        if (kb + 1 < KB) {
            const uint32_t bofs = (uint32_t)((1 - cur) * 64 * 32 * 2);
            async_g2l_b128(ld0 + bofs, bt0 + (kb + 1) * 32);
            async_g2l_b128(ld1 + bofs, bt1 + (kb + 1) * 32);
        }
        const bf16_t* Asrc = (kb < HH / 32) ? ctx : hT;
        const int kloc = (kb < HH / 32) ? kb * 32 : (kb - HH / 32) * 32;

        v16bf a[4], b[4];
#pragma unroll
        for (int s = 0; s < 4; ++s)
            a[s] = load_a_frag(Asrc + (size_t)(mBase + s * 16 + (lane & 15)) * HH + kloc, lane);
#pragma unroll
        for (int t2 = 0; t2 < 4; ++t2)
            b[t2] = load_b_frag(&sB[cur][t2 * 16 + (lane & 15)][0], lane);
#pragma unroll
        for (int s = 0; s < 4; ++s)
#pragma unroll
            for (int t2 = 0; t2 < 4; ++t2) acc[s][t2] = wmma_bf16(a[s], b[t2], acc[s][t2]);
    }

    const int nn = lane & 15, mh = ((lane >> 4) & 1) * 8;
#pragma unroll
    for (int s = 0; s < 4; ++s)
#pragma unroll
        for (int t2 = 0; t2 < 4; ++t2)
#pragma unroll
            for (int r = 0; r < 8; ++r)
                out[(size_t)(mBase + s * 16 + mh + r) * HH + nBase + t2 * 16 + nn] =
                    tanhf(acc[s][t2][r]);
}

// ---------------- Host launcher ----------------

extern "C" void kernel_launch(void* const* d_in, const int* in_sizes, int n_in,
                              void* d_out, int out_size, void* d_ws, size_t ws_size,
                              hipStream_t stream) {
    const float* enc = (const float*)d_in[0];  // [N,L,H]
    const float* ht  = (const float*)d_in[1];  // [N,Q,H]
    const int*   sl  = (const int*)d_in[2];    // [N]
    const float* pt  = (const float*)d_in[3];  // [N]
    const float* Wa  = (const float*)d_in[4];  // [H,H]
    const float* Wc  = (const float*)d_in[5];  // [H,2H]
    float* out = (float*)d_out;                // [N,Q,H]

    char* ws = (char*)d_ws;
    size_t o = 0;
    bf16_t* hTb  = (bf16_t*)(ws + o); o += (size_t)NQ * HH * 2;        // h_t bf16
    bf16_t* WaT  = (bf16_t*)(ws + o); o += (size_t)HH * HH * 2;        // Wa^T bf16
    bf16_t* encb = (bf16_t*)(ws + o); o += (size_t)NB * LL * HH * 2;   // enc bf16
    bf16_t* encT = (bf16_t*)(ws + o); o += (size_t)NB * HH * LP * 2;   // enc^T bf16 (padded)
    bf16_t* Wcb  = (bf16_t*)(ws + o); o += (size_t)HH * 2 * HH * 2;    // Wc bf16
    bf16_t* Tb   = (bf16_t*)(ws + o); o += (size_t)NQ * HH * 2;        // h_t @ Wa
    bf16_t* ctxb = (bf16_t*)(ws + o);                                  // context

    // pack / convert
    {
        int n1 = NQ * HH;
        k_cvt_bf16<<<(n1 + 255) / 256, 256, 0, stream>>>(ht, hTb, n1);
        int n2 = HH * HH;
        k_waT<<<(n2 + 255) / 256, 256, 0, stream>>>(Wa, WaT);
        int n3 = NB * LL * HH;
        k_cvt_bf16<<<(n3 + 255) / 256, 256, 0, stream>>>(enc, encb, n3);
        int n4 = NB * HH * LP;
        k_encT<<<(n4 + 255) / 256, 256, 0, stream>>>(enc, encT);
        int n5 = HH * 2 * HH;
        k_cvt_bf16<<<(n5 + 255) / 256, 256, 0, stream>>>(Wc, Wcb, n5);
    }

    // GEMM1: T = h_t @ W_a   (64 x 16 block-tiles of 256x64)
    k_gemm_TWa<<<(NQ / 256) * (HH / 64), 128, 0, stream>>>(hTb, WaT, Tb);

    // Fused attention (one block per (n, q-tile))
    k_attention<<<NB * (QQ / 16), 128, 0, stream>>>(Tb, encb, encT, sl, pt, ctxb);

    // Output projection + tanh
    k_out<<<(NQ / 256) * (HH / 64), 128, 0, stream>>>(ctxb, hTb, Wcb, out);
}
